// MultiHeadAttentionFLAX_60404420051392
// MI455X (gfx1250) — matrix operand order repaired and checked by reading, not compiled, not run on hardware
//
#include <hip/hip_runtime.h>
#include <hip/hip_bf16.h>

// ---------------------------------------------------------------------------
// MHA forward for MI455X (gfx1250, wave32, WMMA).
// Pipeline: f32->bf16 converts, 3 WMMA GEMMs (QKV proj), flash-attention
// (WMMA scores + online softmax + WMMA PV), WMMA output projection -> f32.
// GEMM waves compute 32x64 tiles: 8 WMMA per 12 b128 loads per K-step.
// ---------------------------------------------------------------------------

typedef __bf16 bf16;
typedef __attribute__((ext_vector_type(16))) __bf16 v16bf;
typedef __attribute__((ext_vector_type(8)))  float  v8f;
typedef unsigned int u32x4 __attribute__((ext_vector_type(4)));
typedef float        f32x4 __attribute__((ext_vector_type(4)));

union Frag {
    v16bf v;
    u32x4 q[2];
};

// A-matrix fragment (16x32 bf16, row-major source):
//   lane L: m = L&15, h = L>>4
//   elems 0..7  = K[h*8 .. h*8+7], elems 8..15 = K[16+h*8 .. 16+h*8+7]
__device__ __forceinline__ v16bf load_a_frag(const bf16* rowp, int h) {
    Frag f;
    const bf16* p = rowp + h * 8;
    f.q[0] = *reinterpret_cast<const u32x4*>(p);
    f.q[1] = *reinterpret_cast<const u32x4*>(p + 16);
    return f.v;
}

// B-matrix fragment (32x16 bf16): lane holds column n = L&15, elems 0..7 =
// K[h*8 .. h*8+7], elems 8..15 = K[16+h*8 ..]. colp points at column base + k0.
__device__ __forceinline__ v16bf load_b_frag(const bf16* colp, int h) {
    Frag f;
    const bf16* p = colp + h * 8;
    f.q[0] = *reinterpret_cast<const u32x4*>(p);
    f.q[1] = *reinterpret_cast<const u32x4*>(p + 16);
    return f.v;
}

__device__ __forceinline__ v8f wmma_bf16(v16bf a, v16bf b, v8f c) {
    return __builtin_amdgcn_wmma_f32_16x16x32_bf16(
        /*neg_a=*/false, a, /*neg_b=*/false, b,
        /*c_mod=*/(short)0, c, /*reuse_a=*/false, /*reuse_b=*/false);
}

// ---------------------------------------------------------------------------
// Stage 0a: f32 -> bf16 elementwise (query)
// ---------------------------------------------------------------------------
__global__ void __launch_bounds__(256)
convert_f32_to_bf16(const float* __restrict__ in, bf16* __restrict__ out) {
    size_t i = ((size_t)blockIdx.x * blockDim.x + threadIdx.x) * 4;
    f32x4 v = *reinterpret_cast<const f32x4*>(in + i);
    out[i + 0] = (bf16)v.x;
    out[i + 1] = (bf16)v.y;
    out[i + 2] = (bf16)v.z;
    out[i + 3] = (bf16)v.w;
}

// ---------------------------------------------------------------------------
// Stage 0b: f32 W[k][n] -> bf16 Wt[n][k]   (LDS-tiled transpose)
// ---------------------------------------------------------------------------
__global__ void __launch_bounds__(256)
transpose_f32_to_bf16(const float* __restrict__ in, bf16* __restrict__ out, int n) {
    __shared__ float tile[32][33];
    const int tx = threadIdx.x, ty = threadIdx.y;
    const int x  = blockIdx.x * 32 + tx;
    const int y0 = blockIdx.y * 32 + ty;
#pragma unroll
    for (int i = 0; i < 32; i += 8)
        tile[ty + i][tx] = in[(size_t)(y0 + i) * n + x];
    __syncthreads();
    const int ox  = blockIdx.y * 32 + tx;
    const int oy0 = blockIdx.x * 32 + ty;
#pragma unroll
    for (int i = 0; i < 32; i += 8)
        out[(size_t)(oy0 + i) * n + ox] = (bf16)tile[tx][ty + i];
}

// ---------------------------------------------------------------------------
// Stage 1/4: GEMM  C[M,N] = A[M,K](bf16) @ Bt[N,K](bf16)^T + bias, epilogue:
//   mode 0: f32 row-major out            (final projection)
//   mode 1: bf16 [b,h,s,d]  * scale      (Q with SCALE folded, K)
//   mode 2: bf16 [b,h,d,s]               (V transposed for PV B-fragments)
// One wave computes a 32x64 tile: 2 A-frags x 4 B-frags -> 8 WMMA per K-step,
// 12 b128 loads per 8 WMMA (1.5 loads/WMMA).
// ---------------------------------------------------------------------------
__global__ void __launch_bounds__(256)
gemm_bf16_kernel(const bf16* __restrict__ A, const bf16* __restrict__ Bt,
                 const float* __restrict__ bias,
                 float* __restrict__ outf, bf16* __restrict__ outb,
                 int M, int K, int N, int mode, float scale)
{
    const int lane = threadIdx.x & 31;
    const int wave = threadIdx.x >> 5;
    const int h  = lane >> 4;
    const int ln = lane & 15;
    const int mtiles = M >> 5;                 // 32-row tiles
    const int tile = blockIdx.x * 8 + wave;
    const int m0 = (tile % mtiles) << 5;
    const int n0 = (tile / mtiles) << 6;

    v8f acc[2][4] = {{{}, {}, {}, {}}, {{}, {}, {}, {}}};
    const bf16* arow0 = A  + (size_t)(m0 +  0 + ln) * K;
    const bf16* arow1 = A  + (size_t)(m0 + 16 + ln) * K;
    const bf16* b0    = Bt + (size_t)(n0 +  0 + ln) * K;
    const bf16* b1    = Bt + (size_t)(n0 + 16 + ln) * K;
    const bf16* b2    = Bt + (size_t)(n0 + 32 + ln) * K;
    const bf16* b3    = Bt + (size_t)(n0 + 48 + ln) * K;

#pragma unroll 4
    for (int k0 = 0; k0 < K; k0 += 32) {
        const v16bf a0 = load_a_frag(arow0 + k0, h);
        const v16bf a1 = load_a_frag(arow1 + k0, h);
        {
            const v16bf b = load_b_frag(b0 + k0, h);
            acc[0][0] = wmma_bf16(a0, b, acc[0][0]);
            acc[1][0] = wmma_bf16(a1, b, acc[1][0]);
        }
        {
            const v16bf b = load_b_frag(b1 + k0, h);
            acc[0][1] = wmma_bf16(a0, b, acc[0][1]);
            acc[1][1] = wmma_bf16(a1, b, acc[1][1]);
        }
        {
            const v16bf b = load_b_frag(b2 + k0, h);
            acc[0][2] = wmma_bf16(a0, b, acc[0][2]);
            acc[1][2] = wmma_bf16(a1, b, acc[1][2]);
        }
        {
            const v16bf b = load_b_frag(b3 + k0, h);
            acc[0][3] = wmma_bf16(a0, b, acc[0][3]);
            acc[1][3] = wmma_bf16(a1, b, acc[1][3]);
        }
    }

#pragma unroll
    for (int i = 0; i < 2; ++i) {
#pragma unroll
        for (int j = 0; j < 4; ++j) {
            const int col = n0 + j * 16 + ln;
            const float bv = bias[col];
#pragma unroll
            for (int r = 0; r < 8; ++r) {
                const int row = m0 + i * 16 + 8 * h + r;
                const float v = (acc[i][j][r] + bv) * scale;
                if (mode == 0) {
                    outf[(size_t)row * N + col] = v;
                } else {
                    const int bb = row >> 11, s = row & 2047;   // S = 2048
                    const int hd = col >> 6,  d = col & 63;     // D = 64
                    const size_t idx = (mode == 1)
                        ? ((size_t)(bb * 16 + hd) * 2048 + s) * 64 + d   // [b,h,s,d]
                        : ((size_t)(bb * 16 + hd) * 64 + d) * 2048 + s;  // [b,h,d,s]
                    outb[idx] = (bf16)v;
                }
            }
        }
    }
}

// ---------------------------------------------------------------------------
// Stage 2/3: flash attention.  One wave owns a 16-query tile of one (b,h).
// Per 32-key step: 4 WMMA (scores, D=64 -> 2 k-steps x 2 key subtiles),
// online softmax with shfl_xor row reductions (lane halves match C-layout),
// P routed C-layout -> A-layout through 1KB of LDS, 4 WMMA for P x V.
// Q arrives pre-scaled by SCALE; K is [b,h,s,d]; V is [b,h,d,s].
// ---------------------------------------------------------------------------
__global__ void __launch_bounds__(256)
flash_attn_kernel(const bf16* __restrict__ Q, const bf16* __restrict__ Kb,
                  const bf16* __restrict__ Vt, bf16* __restrict__ attn)
{
    __shared__ __align__(16) bf16 plds[8][16][32];
    const int lane = threadIdx.x & 31;
    const int wave = threadIdx.x >> 5;
    const int h  = lane >> 4;
    const int ln = lane & 15;

    const int w  = blockIdx.x * 8 + wave;   // 8192 waves total
    const int qt = w & 127;                 // S/16 = 128
    const int hh = (w >> 7) & 15;
    const int b  = w >> 11;
    const int q0 = qt << 4;

    const size_t head = (size_t)(b * 16 + hh) * 2048 * 64;
    const bf16* qrow  = Q  + head + (size_t)(q0 + ln) * 64;
    const bf16* kbase = Kb + head;          // [s][d]
    const bf16* vbase = Vt + head;          // [d][s]

    const v16bf qa0 = load_a_frag(qrow +  0, h);
    const v16bf qa1 = load_a_frag(qrow + 32, h);

    float mrow[8], lrow[8];
    v8f o0 = {}, o1 = {}, o2 = {}, o3 = {};
#pragma unroll
    for (int r = 0; r < 8; ++r) { mrow[r] = -3.0e38f; lrow[r] = 0.0f; }

    for (int kt = 0; kt < 2048; kt += 32) {
        // -------- scores S = Q(16x64) x K^T(64x32)
        const bf16* k0p = kbase + (size_t)(kt      + ln) * 64;
        const bf16* k1p = kbase + (size_t)(kt + 16 + ln) * 64;
        v8f s0 = {}, s1 = {};
        s0 = wmma_bf16(qa0, load_b_frag(k0p +  0, h), s0);
        s0 = wmma_bf16(qa1, load_b_frag(k0p + 32, h), s0);
        s1 = wmma_bf16(qa0, load_b_frag(k1p +  0, h), s1);
        s1 = wmma_bf16(qa1, load_b_frag(k1p + 32, h), s1);

        if (kt + 32 < 2048) {   // pull next K/V tiles toward the WGP
            __builtin_prefetch(kbase + (size_t)(kt + 32 + lane) * 64, 0, 1);
            __builtin_prefetch(vbase + (size_t)lane * 2048 + kt + 32, 0, 1);
            __builtin_prefetch(vbase + (size_t)(32 + lane) * 2048 + kt + 32, 0, 1);
        }

        // -------- online softmax (rows live in lane halves; xor<16 stays put)
        float alpha[8];
#pragma unroll
        for (int r = 0; r < 8; ++r) {
            float mt = fmaxf(s0[r], s1[r]);
            mt = fmaxf(mt, __shfl_xor(mt, 1, 32));
            mt = fmaxf(mt, __shfl_xor(mt, 2, 32));
            mt = fmaxf(mt, __shfl_xor(mt, 4, 32));
            mt = fmaxf(mt, __shfl_xor(mt, 8, 32));
            const float mnew = fmaxf(mrow[r], mt);
            alpha[r] = __expf(mrow[r] - mnew);
            const float p0 = __expf(s0[r] - mnew);
            const float p1 = __expf(s1[r] - mnew);
            s0[r] = p0; s1[r] = p1;
            float ps = p0 + p1;
            ps += __shfl_xor(ps, 1, 32);
            ps += __shfl_xor(ps, 2, 32);
            ps += __shfl_xor(ps, 4, 32);
            ps += __shfl_xor(ps, 8, 32);
            lrow[r] = lrow[r] * alpha[r] + ps;
            mrow[r] = mnew;
        }

        // -------- P: C-layout -> LDS -> A-layout (bf16)
#pragma unroll
        for (int r = 0; r < 8; ++r) {
            plds[wave][8 * h + r][ln]      = (bf16)s0[r];
            plds[wave][8 * h + r][16 + ln] = (bf16)s1[r];
        }
        asm volatile("s_wait_dscnt 0" ::: "memory");   // cross-lane LDS hazard
        const v16bf pa = load_a_frag(&plds[wave][ln][0], h);

        // -------- rescale O, then O += P x V
#pragma unroll
        for (int r = 0; r < 8; ++r) {
            o0[r] *= alpha[r]; o1[r] *= alpha[r];
            o2[r] *= alpha[r]; o3[r] *= alpha[r];
        }
        const bf16* vp0 = vbase + (size_t)( 0 + ln) * 2048 + kt;
        const bf16* vp1 = vbase + (size_t)(16 + ln) * 2048 + kt;
        const bf16* vp2 = vbase + (size_t)(32 + ln) * 2048 + kt;
        const bf16* vp3 = vbase + (size_t)(48 + ln) * 2048 + kt;
        o0 = wmma_bf16(pa, load_b_frag(vp0, h), o0);
        o1 = wmma_bf16(pa, load_b_frag(vp1, h), o1);
        o2 = wmma_bf16(pa, load_b_frag(vp2, h), o2);
        o3 = wmma_bf16(pa, load_b_frag(vp3, h), o3);
    }

    // -------- normalize + store attn as bf16 [b, s, h*64 + d]
    const size_t outbase = (size_t)b * 2048 * 1024;
#pragma unroll
    for (int r = 0; r < 8; ++r) {
        const int srow = q0 + 8 * h + r;
        const float inv = 1.0f / lrow[r];
        const size_t ro = outbase + (size_t)srow * 1024 + hh * 64 + ln;
        attn[ro +  0] = (bf16)(o0[r] * inv);
        attn[ro + 16] = (bf16)(o1[r] * inv);
        attn[ro + 32] = (bf16)(o2[r] * inv);
        attn[ro + 48] = (bf16)(o3[r] * inv);
    }
}

// ---------------------------------------------------------------------------
// Host-side orchestration (all on `stream`, graph-capture safe)
// ---------------------------------------------------------------------------
extern "C" void kernel_launch(void* const* d_in, const int* in_sizes, int n_in,
                              void* d_out, int out_size, void* d_ws, size_t ws_size,
                              hipStream_t stream)
{
    const float* query = (const float*)d_in[0];
    const float* Wq = (const float*)d_in[1];
    const float* bq = (const float*)d_in[2];
    const float* Wk = (const float*)d_in[3];
    const float* bk = (const float*)d_in[4];
    const float* Wv = (const float*)d_in[5];
    const float* bv = (const float*)d_in[6];
    const float* Wo = (const float*)d_in[7];
    const float* bo = (const float*)d_in[8];

    constexpr size_t MB = 1024ull * 1024ull;

    char* ws = (char*)d_ws;
    bf16* Xbf = (bf16*)(ws);                       // query bf16       [8192,1024]
    bf16* Wqt = (bf16*)(ws + 16 * MB);             // Wq^T bf16        [1024,1024]
    bf16* Wkt = (bf16*)(ws + 18 * MB);
    bf16* Wvt = (bf16*)(ws + 20 * MB);
    bf16* Wot = (bf16*)(ws + 22 * MB);
    bf16* Qbf = (bf16*)(ws + 24 * MB);             // Q*SCALE  [b,h,s,d]
    bf16* Kbf = (bf16*)(ws + 40 * MB);             // K        [b,h,s,d]
    bf16* Vtb = (bf16*)(ws + 56 * MB);             // V        [b,h,d,s]
    bf16* Abf = (bf16*)(ws + 72 * MB);             // attn     [8192,1024]
    (void)ws_size; (void)in_sizes; (void)n_in; (void)out_size;

    // 0) converts
    convert_f32_to_bf16<<<8192, 256, 0, stream>>>(query, Xbf);   // 8,388,608 elems
    dim3 tb(32, 8), tg(32, 32);
    transpose_f32_to_bf16<<<tg, tb, 0, stream>>>(Wq, Wqt, 1024);
    transpose_f32_to_bf16<<<tg, tb, 0, stream>>>(Wk, Wkt, 1024);
    transpose_f32_to_bf16<<<tg, tb, 0, stream>>>(Wv, Wvt, 1024);
    transpose_f32_to_bf16<<<tg, tb, 0, stream>>>(Wo, Wot, 1024);

    // 1) QKV projections (8192x1024x1024; 4096 wave-tiles / 8 = 512 blocks)
    const float SCALE = 0.125f;  // 64^-0.5, folded into Q
    gemm_bf16_kernel<<<512, 256, 0, stream>>>(Xbf, Wqt, bq, nullptr, Qbf,
                                              8192, 1024, 1024, 1, SCALE);
    gemm_bf16_kernel<<<512, 256, 0, stream>>>(Xbf, Wkt, bk, nullptr, Kbf,
                                              8192, 1024, 1024, 1, 1.0f);
    gemm_bf16_kernel<<<512, 256, 0, stream>>>(Xbf, Wvt, bv, nullptr, Vtb,
                                              8192, 1024, 1024, 2, 1.0f);

    // 2/3) flash attention: 8192 query-tiles / 8 waves = 1024 blocks
    flash_attn_kernel<<<1024, 256, 0, stream>>>(Qbf, Kbf, Vtb, Abf);

    // 4) output projection -> f32 d_out
    gemm_bf16_kernel<<<512, 256, 0, stream>>>(Abf, Wot, bo, (float*)d_out, nullptr,
                                              8192, 1024, 1024, 0, 1.0f);
}
